// Qwen3MoeDecoderLayer_39548058861860
// MI455X (gfx1250) — matrix operand order repaired and checked by reading, not compiled, not run on hardware
//
#include <hip/hip_runtime.h>
#include <hip/hip_bf16.h>
#include <math.h>

// ---------------- problem constants ----------------
constexpr int CB   = 2;
constexpr int CS   = 1024;
constexpr int CH   = 2048;
constexpr int CNH  = 16;
constexpr int CNKV = 4;
constexpr int CHD  = 128;
constexpr int CE   = 8;
constexpr int CI   = 768;
constexpr int CT   = CB * CS;          // 2048 tokens
constexpr int CDQ  = CNH * CHD;        // 2048
constexpr int CDKV = CNKV * CHD;       // 512

typedef __attribute__((ext_vector_type(16))) __bf16 v16bf;
typedef __attribute__((ext_vector_type(8)))  float  v8f;
typedef __attribute__((ext_vector_type(4)))  unsigned v4u;
typedef __attribute__((ext_vector_type(8)))  unsigned v8u;

union FragU {
    v16bf v;
    uint4 q[2];
    unsigned short s[16];
};

// ---------------- small helpers ----------------
__device__ __forceinline__ unsigned short f2bf_bits(float f) {
    union { float f; unsigned u; } v; v.f = f;
    unsigned u = v.u;
    unsigned r = u + 0x7FFFu + ((u >> 16) & 1u);   // round-to-nearest-even
    return (unsigned short)(r >> 16);
}
__device__ __forceinline__ float bf2f(unsigned short b) {
    union { unsigned u; float f; } v; v.u = ((unsigned)b) << 16;
    return v.f;
}

// Tensor Data Mover: async 2-D tile load Global->LDS (CDNA5 §8).
// Tile: 32 elements/row (64B) x 128 rows, element=2B, row stride = Kelems.
// LDS padding: +4 DWORDs every 16 DWORDs -> row stride 40 halves in LDS.
__device__ __forceinline__ void tdm_load_tile(unsigned lds_addr, const void* gptr, int Kelems) {
    unsigned long long ga = (unsigned long long)gptr;
    v4u g0;
    g0[0] = 1u;                                        // count=1, user descriptor
    g0[1] = lds_addr;                                  // LDS byte address
    g0[2] = (unsigned)ga;                              // global_addr[31:0]
    g0[3] = (unsigned)((ga >> 32) & 0x1FFFFFFu) | (2u << 30); // addr[56:32] | type=2
    v8u g1;
    g1[0] = (1u << 16)        // data_size = 2B
          | (1u << 20)        // pad_enable
          | (3u << 22)        // pad_interval: every 16 DWORDs
          | (3u << 25);       // pad_amount: 4 DWORDs
    g1[1] = ((unsigned)Kelems & 0xFFFFu) << 16;                      // tensor_dim0[15:0]
    g1[2] = (((unsigned)Kelems >> 16) & 0xFFFFu) | (128u << 16);     // dim0 hi | tensor_dim1 lo
    g1[3] = (32u << 16);                                             // tensor_dim1 hi=0 | tile_dim0=32
    g1[4] = 128u;                                                    // tile_dim1=128, tile_dim2=0
    g1[5] = (unsigned)Kelems;                                        // tensor_dim0_stride lo32
    g1[6] = 0u;
    g1[7] = 0u;
    asm volatile("tensor_load_to_lds %0, %1, null, null" :: "s"(g0), "s"(g1) : "memory");
}
__device__ __forceinline__ unsigned lds_off32(const void* p) {
    return (unsigned)(unsigned long long)p;            // flat-LDS aperture keeps offset in [31:0]
}

// ---------------- f32 -> bf16 convert (vectorized) ----------------
__global__ void k_f32_to_bf16_v4(const float4* __restrict__ src,
                                 ushort4* __restrict__ dst, int n4) {
    for (int i = blockIdx.x * blockDim.x + threadIdx.x; i < n4; i += gridDim.x * blockDim.x) {
        float4 f = src[i];
        ushort4 o;
        o.x = f2bf_bits(f.x); o.y = f2bf_bits(f.y);
        o.z = f2bf_bits(f.z); o.w = f2bf_bits(f.w);
        dst[i] = o;
    }
}

// ---------------- V transpose+convert: [T, NKV*HD] f32 -> [B][NKV][HD][S] bf16 ----------------
__global__ void k_transpose_v(const float* __restrict__ vf, unsigned short* __restrict__ vt) {
    int idx = blockIdx.x * blockDim.x + threadIdx.x;     // (bk*CHD + hd)*CS + s
    if (idx >= CB * CNKV * CHD * CS) return;
    int s   = idx & (CS - 1);
    int r   = idx >> 10;                                 // bk*CHD + hd
    int hd  = r & (CHD - 1);
    int bk  = r >> 7;                                    // b*CNKV + kvh
    int b   = bk >> 2;
    int kvh = bk & 3;
    vt[idx] = f2bf_bits(vf[((size_t)(b * CS + s)) * CDKV + kvh * CHD + hd]);
}

// ---------------- RMSNorm (row = ncols) -> bf16 ----------------
__global__ void k_rmsnorm_bf16(const float* __restrict__ x, const float* __restrict__ w,
                               unsigned short* __restrict__ out, int ncols) {
    int row = blockIdx.x;
    const float* xr = x + (size_t)row * ncols;
    float s = 0.f;
    for (int c = threadIdx.x; c < ncols; c += blockDim.x) { float v = xr[c]; s += v * v; }
    __shared__ float red[256];
    red[threadIdx.x] = s;
    __syncthreads();
    for (int off = 128; off > 0; off >>= 1) {
        if ((int)threadIdx.x < off) red[threadIdx.x] += red[threadIdx.x + off];
        __syncthreads();
    }
    float inv = rsqrtf(red[0] / (float)ncols + 1e-6f);
    for (int c = threadIdx.x; c < ncols; c += blockDim.x)
        out[(size_t)row * ncols + c] = f2bf_bits(xr[c] * inv * w[c]);
}

// ---------------- WMMA GEMM: C[M,N] = A[M,K](bf16) x W[N,K]^T(bf16) ----------------
// 256 threads, 8 waves (4x2), block tile 128x128, wave tile 32x64, K-step 32.
// !GATHER: double-buffered TDM (tensor_load_to_lds) pipeline, wave0 drives DMA.
// GATHER : register-pipelined staging with global_prefetch.
// EPI: 0 = store f32 ; 1 = store resid + acc ; 2 = C += acc
template <int EPI, bool GATHER, bool SCATTER>
__global__ void k_gemm_bf16(const unsigned short* __restrict__ A,
                            const unsigned short* __restrict__ W,
                            float* __restrict__ C,
                            const float* __restrict__ resid,
                            const int* __restrict__ rows,
                            const int* __restrict__ cnt,
                            int M, int N, int K) {
    const int nrows = (GATHER || SCATTER) ? cnt[0] : M;
    const int m0 = blockIdx.y * 128;
    if (m0 >= nrows) return;                       // block-uniform exit
    const int n0 = blockIdx.x * 128;

    __shared__ __attribute__((aligned(16))) unsigned short As[2][128][40];
    __shared__ __attribute__((aligned(16))) unsigned short Bs[2][128][40];

    const int tid  = threadIdx.x;
    const int wave = tid >> 5;
    const int lane = tid & 31;
    const int wr   = wave >> 1;                    // 0..3 -> 32 rows each
    const int wc   = wave & 1;                     // 0..1 -> 64 cols each
    const int g    = lane >> 4;
    const int mr   = lane & 15;

    v8f c[2][4];
#pragma unroll
    for (int i = 0; i < 2; ++i)
#pragma unroll
        for (int j = 0; j < 4; ++j)
#pragma unroll
            for (int r = 0; r < 8; ++r) c[i][j][r] = 0.f;

    if (!GATHER) {
        // ---- TDM double-buffered pipeline ----
        const unsigned short* Abase = A + (size_t)m0 * K;       // contiguous rows
        const unsigned short* Bbase = W + (size_t)n0 * K;
        unsigned ldsA[2] = { lds_off32(&As[0][0][0]), lds_off32(&As[1][0][0]) };
        unsigned ldsB[2] = { lds_off32(&Bs[0][0][0]), lds_off32(&Bs[1][0][0]) };

        if (wave == 0) {
            tdm_load_tile(ldsA[0], Abase, K);
            tdm_load_tile(ldsB[0], Bbase, K);
        }
        for (int kk = 0; kk < K; kk += 32) {
            const int cur = (kk >> 5) & 1;
            if (wave == 0) {
                if (kk + 32 < K) {
                    tdm_load_tile(ldsA[cur ^ 1], Abase + kk + 32, K);
                    tdm_load_tile(ldsB[cur ^ 1], Bbase + kk + 32, K);
                    __builtin_amdgcn_s_wait_tensorcnt(2);   // current pair landed
                } else {
                    __builtin_amdgcn_s_wait_tensorcnt(0);
                }
            }
            __syncthreads();

            FragU ua[2], ub[4];
#pragma unroll
            for (int i = 0; i < 2; ++i) {
                const unsigned short* ar = &As[cur][wr * 32 + i * 16 + mr][0];
                ua[i].q[0] = *(const uint4*)(ar + g * 8);
                ua[i].q[1] = *(const uint4*)(ar + 16 + g * 8);
            }
#pragma unroll
            for (int j = 0; j < 4; ++j) {
                const unsigned short* br = &Bs[cur][wc * 64 + j * 16 + mr][0];
                ub[j].q[0] = *(const uint4*)(br + g * 8);
                ub[j].q[1] = *(const uint4*)(br + 16 + g * 8);
            }
#pragma unroll
            for (int i = 0; i < 2; ++i)
#pragma unroll
                for (int j = 0; j < 4; ++j)
                    c[i][j] = __builtin_amdgcn_wmma_f32_16x16x32_bf16(
                        false, ua[i].v, false, ub[j].v, (short)0, c[i][j], false, false);
            __syncthreads();   // buffer reusable by the issue two steps ahead
        }
    } else {
        // ---- gathered-A path: register-pipelined staging ----
        int lr[2], lcq[2], arow[2];
#pragma unroll
        for (int p = 0; p < 2; ++p) {
            int li = tid + p * 256;                    // 0..511
            lr[p]  = li >> 2;
            lcq[p] = li & 3;
            int m = m0 + lr[p];
            int mm = (m < nrows) ? m : (nrows - 1);
            arow[p] = rows[mm];
        }
        uint4 pva[2], pvb[2];
#pragma unroll
        for (int p = 0; p < 2; ++p) {
            pva[p] = *(const uint4*)(A + (size_t)arow[p] * K + lcq[p] * 8);
            pvb[p] = *(const uint4*)(W + (size_t)(n0 + lr[p]) * K + lcq[p] * 8);
        }
        for (int kk = 0; kk < K; kk += 32) {
#pragma unroll
            for (int p = 0; p < 2; ++p) {
                *(uint4*)(&As[0][lr[p]][lcq[p] * 8]) = pva[p];
                *(uint4*)(&Bs[0][lr[p]][lcq[p] * 8]) = pvb[p];
            }
            __syncthreads();
            if (kk + 32 < K) {                          // prefetch next slice (overlaps WMMAs)
#pragma unroll
                for (int p = 0; p < 2; ++p) {
                    pva[p] = *(const uint4*)(A + (size_t)arow[p] * K + kk + 32 + lcq[p] * 8);
                    pvb[p] = *(const uint4*)(W + (size_t)(n0 + lr[p]) * K + kk + 32 + lcq[p] * 8);
                }
                if (kk + 64 < K) {                      // L2 prefetch one more slice ahead
                    __builtin_prefetch(A + (size_t)arow[0] * K + kk + 64 + lcq[0] * 8, 0, 1);
                    __builtin_prefetch(W + (size_t)(n0 + lr[0]) * K + kk + 64 + lcq[0] * 8, 0, 1);
                }
            }
            FragU ua[2], ub[4];
#pragma unroll
            for (int i = 0; i < 2; ++i) {
                const unsigned short* ar = &As[0][wr * 32 + i * 16 + mr][0];
                ua[i].q[0] = *(const uint4*)(ar + g * 8);
                ua[i].q[1] = *(const uint4*)(ar + 16 + g * 8);
            }
#pragma unroll
            for (int j = 0; j < 4; ++j) {
                const unsigned short* br = &Bs[0][wc * 64 + j * 16 + mr][0];
                ub[j].q[0] = *(const uint4*)(br + g * 8);
                ub[j].q[1] = *(const uint4*)(br + 16 + g * 8);
            }
#pragma unroll
            for (int i = 0; i < 2; ++i)
#pragma unroll
                for (int j = 0; j < 4; ++j)
                    c[i][j] = __builtin_amdgcn_wmma_f32_16x16x32_bf16(
                        false, ua[i].v, false, ub[j].v, (short)0, c[i][j], false, false);
            __syncthreads();
        }
    }

#pragma unroll
    for (int i = 0; i < 2; ++i)
#pragma unroll
        for (int j = 0; j < 4; ++j) {
            int n = n0 + wc * 64 + j * 16 + mr;
#pragma unroll
            for (int r = 0; r < 8; ++r) {
                int m = m0 + wr * 32 + i * 16 + g * 8 + r;
                float v = c[i][j][r];
                if (SCATTER) {
                    if (m < nrows) {
                        size_t idx = (size_t)rows[m] * N + n;
                        C[idx] = C[idx] + v;
                    }
                } else if (GATHER) {
                    if (m < nrows) C[(size_t)m * N + n] = v;
                } else {
                    size_t idx = (size_t)m * N + n;
                    if (EPI == 0)      C[idx] = v;
                    else if (EPI == 1) C[idx] = resid[idx] + v;
                    else               C[idx] = C[idx] + v;
                }
            }
        }
}

// ---------------- per-head RMSNorm + RoPE -> bf16 ----------------
__global__ void k_rope_norm(const float* __restrict__ src, const float* __restrict__ nw,
                            const float* __restrict__ cosb, const float* __restrict__ sinb,
                            unsigned short* __restrict__ dst, int nheads) {
    int bid = blockIdx.x;
    int t = bid / nheads, h = bid % nheads;
    int i = threadIdx.x;
    const float* row = src + ((size_t)t * nheads + h) * CHD;
    float v = row[i];
    __shared__ float red[128];
    __shared__ float qn[128];
    red[i] = v * v;
    __syncthreads();
    for (int off = 64; off > 0; off >>= 1) {
        if (i < off) red[i] += red[i + off];
        __syncthreads();
    }
    float inv = rsqrtf(red[0] / (float)CHD + 1e-6f);
    qn[i] = v * inv * nw[i];
    __syncthreads();
    float rot = (i < 64) ? -qn[i + 64] : qn[i - 64];
    float c = cosb[(size_t)t * CHD + i];
    float s = sinb[(size_t)t * CHD + i];
    dst[((size_t)t * nheads + h) * CHD + i] = f2bf_bits(qn[i] * c + rot * s);
}

// ---------------- flash attention (1 wave per (b,h,16-query tile)) ----------------
// Q,K: [T, nheads*HD] bf16 ; Vt: [B][NKV][HD][S] bf16 (pre-transposed)
__global__ void k_flash_attn(const unsigned short* __restrict__ Qb,
                             const unsigned short* __restrict__ Kb,
                             const unsigned short* __restrict__ Vt,
                             unsigned short* __restrict__ Ob) {
    const int QT = CS / 16;
    int bid = blockIdx.x;
    int qt  = bid % QT;
    int bh  = bid / QT;
    int h   = bh % CNH;
    int b   = bh / CNH;
    int kvh = h / (CNH / CNKV);
    int lane = threadIdx.x, g = lane >> 4, mr = lane & 15;
    int qbase = qt * 16;
    const float scale = 0.08838834764831845f;   // 128^-0.5

    __shared__ __attribute__((aligned(16))) unsigned short qs[16][136]; // padded stride
    __shared__ __attribute__((aligned(16))) unsigned short pl[16][32];

    // stage Q tile (16x128) into LDS
#pragma unroll
    for (int p = 0; p < 8; ++p) {
        int li = lane + p * 32;                  // 0..255
        int r  = li >> 4, cq = li & 15;
        *(uint4*)(&qs[r][cq * 8]) =
            *(const uint4*)(Qb + (((size_t)(b * CS + qbase + r)) * CNH + h) * CHD + cq * 8);
    }
    __syncthreads();

    float mstate[8], lstate[8];
    v8f o[8];
#pragma unroll
    for (int r = 0; r < 8; ++r) { mstate[r] = -1e30f; lstate[r] = 0.f; }
#pragma unroll
    for (int j = 0; j < 8; ++j)
#pragma unroll
        for (int r = 0; r < 8; ++r) o[j][r] = 0.f;

    const unsigned short* vbase = Vt + ((size_t)(b * CNKV + kvh)) * CHD * CS;

    int kend = qbase + 16;   // exclusive causal limit
    for (int kb = 0; kb < kend; kb += 32) {
        v8f sc[2];
#pragma unroll
        for (int x = 0; x < 2; ++x)
#pragma unroll
            for (int r = 0; r < 8; ++r) sc[x][r] = 0.f;
#pragma unroll
        for (int kc = 0; kc < 4; ++kc) {
            FragU qa;
            qa.q[0] = *(const uint4*)(&qs[mr][kc * 32 + g * 8]);
            qa.q[1] = *(const uint4*)(&qs[mr][kc * 32 + 16 + g * 8]);
#pragma unroll
            for (int x = 0; x < 2; ++x) {
                const unsigned short* krow =
                    Kb + (((size_t)(b * CS + kb + x * 16 + mr)) * CNKV + kvh) * CHD + kc * 32;
                FragU u;
                u.q[0] = *(const uint4*)(krow + g * 8);
                u.q[1] = *(const uint4*)(krow + 16 + g * 8);
                sc[x] = __builtin_amdgcn_wmma_f32_16x16x32_bf16(
                    false, qa.v, false, u.v, (short)0, sc[x], false, false);
            }
        }
#pragma unroll
        for (int r = 0; r < 8; ++r) {
            int qrow_i = qbase + g * 8 + r;
#pragma unroll
            for (int x = 0; x < 2; ++x) {
                int key = kb + x * 16 + mr;
                float sv = sc[x][r] * scale;
                sc[x][r] = (key <= qrow_i) ? sv : -1e30f;
            }
            float mx = fmaxf(sc[0][r], sc[1][r]);
#pragma unroll
            for (int off = 1; off < 16; off <<= 1) mx = fmaxf(mx, __shfl_xor(mx, off, 32));
            float mnew = fmaxf(mstate[r], mx);
            float corr = __expf(mstate[r] - mnew);
            float p0 = __expf(sc[0][r] - mnew);
            float p1 = __expf(sc[1][r] - mnew);
            float psum = p0 + p1;
#pragma unroll
            for (int off = 1; off < 16; off <<= 1) psum += __shfl_xor(psum, off, 32);
            lstate[r] = lstate[r] * corr + psum;
            mstate[r] = mnew;
            pl[g * 8 + r][mr]      = f2bf_bits(p0);
            pl[g * 8 + r][16 + mr] = f2bf_bits(p1);
#pragma unroll
            for (int j = 0; j < 8; ++j) o[j][r] = o[j][r] * corr;
        }
        __syncthreads();   // P tile visible in LDS

        FragU up;
        up.q[0] = *(const uint4*)(&pl[mr][g * 8]);
        up.q[1] = *(const uint4*)(&pl[mr][16 + g * 8]);
#pragma unroll
        for (int j = 0; j < 8; ++j) {
            const unsigned short* vrow = vbase + (size_t)(j * 16 + mr) * CS + kb;
            FragU uv;
            uv.q[0] = *(const uint4*)(vrow + g * 8);
            uv.q[1] = *(const uint4*)(vrow + 16 + g * 8);
            o[j] = __builtin_amdgcn_wmma_f32_16x16x32_bf16(
                false, up.v, false, uv.v, (short)0, o[j], false, false);
        }
        __syncthreads();
    }

#pragma unroll
    for (int r = 0; r < 8; ++r) {
        float rinv = 1.f / lstate[r];
        int qrow_i = qbase + g * 8 + r;
#pragma unroll
        for (int j = 0; j < 8; ++j)
            Ob[(((size_t)(b * CS + qrow_i)) * CNH + h) * CHD + j * 16 + mr] =
                f2bf_bits(o[j][r] * rinv);
    }
}

// ---------------- routing: gate logits + softmax + top-2 + expert token lists ----------------
__global__ void k_zero_route(int* __restrict__ counts) {
    if ((int)threadIdx.x < CE) counts[threadIdx.x] = 0;
}

__global__ void k_gate_topk(const unsigned short* __restrict__ x2,
                            const float* __restrict__ gw,
                            int* __restrict__ counts,
                            int* __restrict__ elist,
                            float* __restrict__ wlist) {
    int t = blockIdx.x;
    int tid = threadIdx.x;
    float acc[CE];
#pragma unroll
    for (int e = 0; e < CE; ++e) acc[e] = 0.f;
    for (int c = tid; c < CH; c += 128) {
        float xv = bf2f(x2[(size_t)t * CH + c]);
#pragma unroll
        for (int e = 0; e < CE; ++e) acc[e] += xv * gw[(size_t)e * CH + c];
    }
    __shared__ float red[CE][128];
#pragma unroll
    for (int e = 0; e < CE; ++e) red[e][tid] = acc[e];
    __syncthreads();
    for (int off = 64; off > 0; off >>= 1) {
        if (tid < off)
#pragma unroll
            for (int e = 0; e < CE; ++e) red[e][tid] += red[e][tid + off];
        __syncthreads();
    }
    if (tid == 0) {
        float lg[CE];
        float mx = -1e30f;
#pragma unroll
        for (int e = 0; e < CE; ++e) { lg[e] = red[e][0]; mx = fmaxf(mx, lg[e]); }
        float den = 0.f, pr[CE];
#pragma unroll
        for (int e = 0; e < CE; ++e) { pr[e] = __expf(lg[e] - mx); den += pr[e]; }
#pragma unroll
        for (int e = 0; e < CE; ++e) pr[e] /= den;
        int i0 = 0;
#pragma unroll
        for (int e = 1; e < CE; ++e) if (pr[e] > pr[i0]) i0 = e;
        int i1 = (i0 == 0) ? 1 : 0;
#pragma unroll
        for (int e = 0; e < CE; ++e) if (e != i0 && pr[e] > pr[i1]) i1 = e;
        float s = pr[i0] + pr[i1];
        int s0 = atomicAdd(&counts[i0], 1);
        elist[(size_t)i0 * CT + s0] = t;
        wlist[(size_t)i0 * CT + s0] = pr[i0] / s;
        int s1 = atomicAdd(&counts[i1], 1);
        elist[(size_t)i1 * CT + s1] = t;
        wlist[(size_t)i1 * CT + s1] = pr[i1] / s;
    }
}

// ---------------- SiLU(g)*u*w -> bf16 over compacted expert rows ----------------
// grid: (CI/256, CT); blockIdx.y = slot
__global__ void k_silu_mul(const float* __restrict__ g, const float* __restrict__ u,
                           const float* __restrict__ wslot, const int* __restrict__ cnt,
                           unsigned short* __restrict__ act) {
    int slot = blockIdx.y;
    if (slot >= cnt[0]) return;
    int i = slot * CI + blockIdx.x * blockDim.x + threadIdx.x;
    float w  = wslot[slot];
    float gv = g[i];
    float sl = gv / (1.f + __expf(-gv));
    act[i] = f2bf_bits(sl * u[i] * w);
}

// ---------------- launch ----------------
extern "C" void kernel_launch(void* const* d_in, const int* in_sizes, int n_in,
                              void* d_out, int out_size, void* d_ws, size_t ws_size,
                              hipStream_t stream) {
    (void)in_sizes; (void)n_in; (void)out_size; (void)ws_size;

    const float* hidden = (const float*)d_in[0];
    const float* cosb   = (const float*)d_in[2];
    const float* sinb   = (const float*)d_in[3];
    const float* w_q    = (const float*)d_in[5];
    const float* w_k    = (const float*)d_in[6];
    const float* w_v    = (const float*)d_in[7];
    const float* w_o    = (const float*)d_in[8];
    const float* q_nw   = (const float*)d_in[9];
    const float* k_nw   = (const float*)d_in[10];
    const float* ln1_w  = (const float*)d_in[11];
    const float* ln2_w  = (const float*)d_in[12];
    const float* gate_w = (const float*)d_in[13];
    const float* w_ge   = (const float*)d_in[14];
    const float* w_ue   = (const float*)d_in[15];
    const float* w_de   = (const float*)d_in[16];
    float* out = (float*)d_out;

    // workspace carving
    char* ws = (char*)d_ws;
    size_t off = 0;
    auto carve = [&](size_t bytes) -> void* {
        void* p = ws + off;
        off += bytes;
        off = (off + 255) & ~(size_t)255;
        return p;
    };
    unsigned short* wq_bf  = (unsigned short*)carve((size_t)CDQ * CH * 2);
    unsigned short* wk_bf  = (unsigned short*)carve((size_t)CDKV * CH * 2);
    unsigned short* wv_bf  = (unsigned short*)carve((size_t)CDKV * CH * 2);
    unsigned short* wo_bf  = (unsigned short*)carve((size_t)CH * CDQ * 2);
    unsigned short* wg_bf  = (unsigned short*)carve((size_t)CE * CI * CH * 2);
    unsigned short* wu_bf  = (unsigned short*)carve((size_t)CE * CI * CH * 2);
    unsigned short* wd_bf  = (unsigned short*)carve((size_t)CE * CH * CI * 2);
    unsigned short* x_bf   = (unsigned short*)carve((size_t)CT * CH * 2);
    float*          q_f    = (float*)carve((size_t)CT * CDQ * 4);
    float*          k_f    = (float*)carve((size_t)CT * CDKV * 4);
    float*          v_f    = (float*)carve((size_t)CT * CDKV * 4);
    unsigned short* q_bf   = (unsigned short*)carve((size_t)CT * CDQ * 2);
    unsigned short* k_bf   = (unsigned short*)carve((size_t)CT * CDKV * 2);
    unsigned short* vt_bf  = (unsigned short*)carve((size_t)CT * CDKV * 2);
    unsigned short* ctx_bf = (unsigned short*)carve((size_t)CT * CDQ * 2);
    unsigned short* x2_bf  = (unsigned short*)carve((size_t)CT * CH * 2);
    int*            counts = (int*)carve((size_t)CE * 4);
    int*            elist  = (int*)carve((size_t)CE * CT * 4);
    float*          wlist  = (float*)carve((size_t)CE * CT * 4);
    float*          g_f    = (float*)carve((size_t)CT * CI * 4);
    float*          u_f    = (float*)carve((size_t)CT * CI * 4);
    unsigned short* act_bf = (unsigned short*)carve((size_t)CT * CI * 2);

    auto conv = [&](const float* s, unsigned short* d, size_t n) {
        int n4 = (int)(n / 4);
        int blocks = (n4 + 255) / 256;
        k_f32_to_bf16_v4<<<blocks, 256, 0, stream>>>((const float4*)s, (ushort4*)d, n4);
    };
    // 1) weights -> bf16 once (~96 MB bf16 total: L2-resident for the GEMMs)
    conv(w_q, wq_bf, (size_t)CDQ * CH);
    conv(w_k, wk_bf, (size_t)CDKV * CH);
    conv(w_v, wv_bf, (size_t)CDKV * CH);
    conv(w_o, wo_bf, (size_t)CH * CDQ);
    conv(w_ge, wg_bf, (size_t)CE * CI * CH);
    conv(w_ue, wu_bf, (size_t)CE * CI * CH);
    conv(w_de, wd_bf, (size_t)CE * CH * CI);

    // 2) ln1
    k_rmsnorm_bf16<<<CT, 256, 0, stream>>>(hidden, ln1_w, x_bf, CH);

    // 3) QKV projections (WMMA + TDM)
    k_gemm_bf16<0, false, false><<<dim3(CDQ / 128, CT / 128), 256, 0, stream>>>(
        x_bf, wq_bf, q_f, nullptr, nullptr, nullptr, CT, CDQ, CH);
    k_gemm_bf16<0, false, false><<<dim3(CDKV / 128, CT / 128), 256, 0, stream>>>(
        x_bf, wk_bf, k_f, nullptr, nullptr, nullptr, CT, CDKV, CH);
    k_gemm_bf16<0, false, false><<<dim3(CDKV / 128, CT / 128), 256, 0, stream>>>(
        x_bf, wv_bf, v_f, nullptr, nullptr, nullptr, CT, CDKV, CH);

    // 4) q/k-norm + RoPE -> bf16 ; V -> transposed bf16
    k_rope_norm<<<CT * CNH, 128, 0, stream>>>(q_f, q_nw, cosb, sinb, q_bf, CNH);
    k_rope_norm<<<CT * CNKV, 128, 0, stream>>>(k_f, k_nw, cosb, sinb, k_bf, CNKV);
    {
        int n = CT * CDKV;
        k_transpose_v<<<(n + 255) / 256, 256, 0, stream>>>(v_f, vt_bf);
    }

    // 5) flash attention (WMMA)
    k_flash_attn<<<CB * CNH * (CS / 16), 32, 0, stream>>>(q_bf, k_bf, vt_bf, ctx_bf);

    // 6) O projection + residual -> d_out
    k_gemm_bf16<1, false, false><<<dim3(CH / 128, CT / 128), 256, 0, stream>>>(
        ctx_bf, wo_bf, out, hidden, nullptr, nullptr, CT, CH, CDQ);

    // 7) ln2 + top-2 routing (builds per-expert compact token lists)
    k_rmsnorm_bf16<<<CT, 256, 0, stream>>>(out, ln2_w, x2_bf, CH);
    k_zero_route<<<1, 32, 0, stream>>>(counts);
    k_gate_topk<<<CT, 128, 0, stream>>>(x2_bf, gate_w, counts, elist, wlist);

    // 8) routed MoE experts (WMMA): gather -> gate/up -> SiLU*up*w -> scatter-add down
    for (int e = 0; e < CE; ++e) {
        const unsigned short* wg_e = wg_bf + (size_t)e * CI * CH;
        const unsigned short* wu_e = wu_bf + (size_t)e * CI * CH;
        const unsigned short* wd_e = wd_bf + (size_t)e * CH * CI;
        const int* rows_e = elist + (size_t)e * CT;
        const int* cnt_e  = counts + e;
        k_gemm_bf16<0, true, false><<<dim3(CI / 128, CT / 128), 256, 0, stream>>>(
            x2_bf, wg_e, g_f, nullptr, rows_e, cnt_e, CT, CI, CH);
        k_gemm_bf16<0, true, false><<<dim3(CI / 128, CT / 128), 256, 0, stream>>>(
            x2_bf, wu_e, u_f, nullptr, rows_e, cnt_e, CT, CI, CH);
        k_silu_mul<<<dim3(CI / 256, CT), 256, 0, stream>>>(g_f, u_f, wlist + (size_t)e * CT,
                                                           cnt_e, act_bf);
        k_gemm_bf16<2, false, true><<<dim3(CH / 128, CT / 128), 256, 0, stream>>>(
            act_bf, wd_e, out, nullptr, rows_e, cnt_e, CT, CH, CI);
    }
}
